// MultiHeadSelfAttention_31568009626306
// MI455X (gfx1250) — matrix operand order repaired and checked by reading, not compiled, not run on hardware
//
#include <hip/hip_runtime.h>

// ---------------------------------------------------------------------------
// MHSA for MI455X (gfx1250): bf16 WMMA 16x16x32 + TDM (tensor_load_to_lds)
// double-buffered staging. fp32 accumulation, fp32 softmax statistics.
// ~69 GFLOP vs ~50MB traffic (all L2-resident) -> purely WMMA-throughput
// bound; design goal is keeping matrix pipes fed with minimal barrier cost.
// ---------------------------------------------------------------------------

typedef __bf16 bf16;
typedef __attribute__((ext_vector_type(16))) __bf16 v16bf;
typedef __attribute__((ext_vector_type(8)))  __bf16 v8bf;
typedef __attribute__((ext_vector_type(4)))  __bf16 v4bf;
typedef __attribute__((ext_vector_type(8)))  float  v8f;
typedef __attribute__((ext_vector_type(4)))  float  v4f;
typedef __attribute__((ext_vector_type(4)))  unsigned int u32x4;
typedef __attribute__((ext_vector_type(8)))  int i32x8;
typedef __attribute__((ext_vector_type(4)))  int i32x4;

#define D_MODEL   1024
#define NUM_HEADS 16
#define HEAD_DIM  64
#define SEQ       2048
#define BATCH     2
#define ROWS      (BATCH * SEQ)   // 4096

__device__ __forceinline__ v16bf mk16(v8bf lo, v8bf hi) {
  v16bf r;
#pragma unroll
  for (int i = 0; i < 8; ++i) { r[i] = lo[i]; r[i + 8] = hi[i]; }
  return r;
}

__device__ __forceinline__ v8f wmma_bf16(v16bf a, v16bf b, v8f c) {
  return __builtin_amdgcn_wmma_f32_16x16x32_bf16(
      false, a, false, b, (short)0, c, false, false);
}

// --- permlane16-based reductions within each 16-lane row (VALU, no LDS) ----
__device__ __forceinline__ float permx16(float v, unsigned lo, unsigned hi) {
  unsigned u = __builtin_bit_cast(unsigned, v);
  unsigned r = __builtin_amdgcn_permlane16(u, u, lo, hi, false, false);
  return __builtin_bit_cast(float, r);
}
__device__ __forceinline__ float redmax16(float v) {
  v = fmaxf(v, permx16(v, 0x67452301u, 0xEFCDAB89u));  // xor 1
  v = fmaxf(v, permx16(v, 0x54761032u, 0xDCFE98BAu));  // xor 2
  v = fmaxf(v, permx16(v, 0x32107654u, 0xBA98FEDCu));  // xor 4
  v = fmaxf(v, permx16(v, 0xFEDCBA98u, 0x76543210u));  // xor 8
  return v;
}
__device__ __forceinline__ float redsum16(float v) {
  v += permx16(v, 0x67452301u, 0xEFCDAB89u);
  v += permx16(v, 0x54761032u, 0xDCFE98BAu);
  v += permx16(v, 0x32107654u, 0xBA98FEDCu);
  v += permx16(v, 0xFEDCBA98u, 0x76543210u);
  return v;
}

// ---------------------------------------------------------------------------
// TDM: 2D tile DMA global -> LDS with per-row LDS padding.
//   tile_d0 elems per row (contiguous), tile_d1 rows, row stride in elems,
//   pad codes: interval 3 -> every 16 DW (64B), 4 -> every 32 DW (128B);
//              amount  3 -> 4 DW (16B).
// ---------------------------------------------------------------------------
__device__ __forceinline__ void tdm_load_2d(unsigned lds_addr, const void* gaddr,
                                            unsigned tile_d0, unsigned tile_d1,
                                            unsigned stride_elems,
                                            unsigned pad_interval, unsigned pad_amount) {
  unsigned long long ga = (unsigned long long)(__UINTPTR_TYPE__)gaddr;
  u32x4 g0;
  g0[0] = 1u;                                            // count=1 (user mode)
  g0[1] = lds_addr;                                      // LDS byte address
  g0[2] = (unsigned)(ga & 0xFFFFFFFFu);                  // global_addr[31:0]
  g0[3] = (unsigned)((ga >> 32) & 0x01FFFFFFu) | (2u << 30);  // [56:32] | type=2
  i32x8 g1;
  g1[0] = (int)((1u << 16)                     // data_size = 2B (bf16)
                | (1u << 20)                   // pad_enable
                | (pad_interval << 22) | (pad_amount << 25));
  g1[1] = (int)((tile_d0 & 0xFFFFu) << 16);    // tensor_dim0 lo16 (== tile_d0)
  g1[2] = (int)((tile_d1 & 0xFFFFu) << 16);    // tensor_dim0 hi=0 | tensor_dim1 lo16
  g1[3] = (int)((tile_d0 & 0xFFFFu) << 16);    // tensor_dim1 hi=0 | tile_dim0
  g1[4] = (int)(tile_d1 & 0xFFFFu);            // tile_dim1 | tile_dim2=0
  g1[5] = (int)stride_elems;                   // tensor_dim0_stride lo32
  g1[6] = 0;                                   // stride hi16 | dim1_stride lo16
  g1[7] = 0;
  i32x4 z4 = {0, 0, 0, 0};
  i32x8 z8 = {0, 0, 0, 0, 0, 0, 0, 0};
  __builtin_amdgcn_tensor_load_to_lds(g0, g1, z4, z4, z8, 0);
}

// ---------------------------------------------------------------------------
// fp32 -> bf16 convert
// ---------------------------------------------------------------------------
__global__ __launch_bounds__(256)
void cvt_f32_bf16(const float* __restrict__ in, bf16* __restrict__ out, int n) {
  int i = (blockIdx.x * 256 + threadIdx.x) * 4;
  if (i + 3 < n) {
    v4f v = *(const v4f*)(in + i);
    v4bf o;
#pragma unroll
    for (int j = 0; j < 4; ++j) o[j] = (bf16)v[j];
    *(v4bf*)(out + i) = o;
  }
}

// ---------------------------------------------------------------------------
// GEMM: C[M,N] = A[M,K] * B[K,N] (bf16 rm). WG=256 thr = 8 waves.
// WG tile 128x128, BK=32, wave tile 32x64 (2x4 accums -> 8 WMMA/iter).
// A tile staged by TDM (padded rows, stride 40); B tile staged manually
// transposed [N][K] (stride 40). Double-buffered, 1 barrier per K-iter.
// ---------------------------------------------------------------------------
template <bool OUT_F32>
__global__ __launch_bounds__(256)
void gemm_bf16_kernel(const bf16* __restrict__ A, const bf16* __restrict__ B,
                      bf16* __restrict__ Cb, float* __restrict__ Cf,
                      int M, int N, int K) {
  __shared__ bf16 Al[2][128 * 40];
  __shared__ bf16 Bt[2][128 * 40];

  const int tid  = threadIdx.x;
  const int lane = tid & 31;
  const int w    = tid >> 5;
  const int wr   = w & 3;    // 4 row blocks of 32
  const int wc   = w >> 2;   // 2 col blocks of 64
  const int lr   = lane & 15;
  const int sel  = lane >> 4;

  const int bm0 = blockIdx.y * 128;
  const int bn0 = blockIdx.x * 128;

  v8f acc[2][4] = {};

  const int nk = K / 32;

  // ---- prologue: stage buffer 0
  if (w == 0)
    tdm_load_2d((unsigned)(__UINTPTR_TYPE__)&Al[0][0],
                &A[(size_t)bm0 * K], 32, 128, (unsigned)K, 3u, 3u);
  {
#pragma unroll
    for (int it = 0; it < 2; ++it) {
      int idx = tid + it * 256;            // 0..511
      int k   = idx >> 4;                  // 0..31
      int n8  = (idx & 15) * 8;            // 0..120
      v8bf v  = *(const v8bf*)&B[(size_t)k * N + bn0 + n8];
#pragma unroll
      for (int j = 0; j < 8; ++j) Bt[0][(n8 + j) * 40 + k] = v[j];
    }
  }

  for (int kt = 0; kt < nk; ++kt) {
    const int cur = kt & 1, nxt = cur ^ 1;
    if (w == 0) __builtin_amdgcn_s_wait_tensorcnt(0);  // Al[cur] DMA complete
    __syncthreads();                                   // all staging visible

    const bool pre = (kt + 1) < nk;
    const int k0n  = (kt + 1) * 32;

    // kick next A-tile DMA into the other buffer while we compute
    if (pre && w == 0)
      tdm_load_2d((unsigned)(__UINTPTR_TYPE__)&Al[nxt][0],
                  &A[(size_t)bm0 * K + k0n], 32, 128, (unsigned)K, 3u, 3u);

    // issue next B-tile global loads early (latency hidden by WMMAs)
    v8bf bv[2];
    if (pre) {
#pragma unroll
      for (int it = 0; it < 2; ++it) {
        int idx = tid + it * 256;
        int k   = idx >> 4;
        int n8  = (idx & 15) * 8;
        bv[it]  = *(const v8bf*)&B[(size_t)(k0n + k) * N + bn0 + n8];
      }
    }

    // fragments
    v16bf af[2], bfr[4];
#pragma unroll
    for (int r = 0; r < 2; ++r) {
      const bf16* p = &Al[cur][(wr * 32 + r * 16 + lr) * 40];
      af[r] = mk16(*(const v8bf*)(p + sel * 8), *(const v8bf*)(p + 16 + sel * 8));
    }
#pragma unroll
    for (int c = 0; c < 4; ++c) {
      const bf16* p = &Bt[cur][(wc * 64 + c * 16 + lr) * 40 + sel * 8];
      bfr[c] = mk16(*(const v8bf*)p, *(const v8bf*)(p + 16));
    }
#pragma unroll
    for (int r = 0; r < 2; ++r)
#pragma unroll
      for (int c = 0; c < 4; ++c)
        acc[r][c] = wmma_bf16(af[r], bfr[c], acc[r][c]);

    // commit next B tile (transposed) into the other buffer
    if (pre) {
#pragma unroll
      for (int it = 0; it < 2; ++it) {
        int idx = tid + it * 256;
        int k   = idx >> 4;
        int n8  = (idx & 15) * 8;
#pragma unroll
        for (int j = 0; j < 8; ++j) Bt[nxt][(n8 + j) * 40 + k] = bv[it][j];
      }
    }
  }

  // C/D layout: VGPR i -> row i (lanes 0-15) / i+8 (lanes 16-31); col = lane%16
#pragma unroll
  for (int r = 0; r < 2; ++r)
#pragma unroll
    for (int c = 0; c < 4; ++c) {
      int col = bn0 + wc * 64 + c * 16 + lr;
#pragma unroll
      for (int i = 0; i < 8; ++i) {
        int row = bm0 + wr * 32 + r * 16 + i + sel * 8;
        if (OUT_F32) Cf[(size_t)row * N + col] = acc[r][c][i];
        else         Cb[(size_t)row * N + col] = (bf16)acc[r][c][i];
      }
    }
}

// ---------------------------------------------------------------------------
// Flash attention: grid (T/128, B*H). 8 waves; wave owns 16 query rows.
// 32-key chunks: K tile via TDM (no transpose needed for Q@K^T B-frags),
// V tile manually transposed. Double-buffered, 1 barrier per chunk.
// P (C-layout) bounced through LDS into A-layout for P@V.
// ---------------------------------------------------------------------------
__global__ __launch_bounds__(256)
void attn_kernel(const bf16* __restrict__ Q, const bf16* __restrict__ Kb,
                 const bf16* __restrict__ Vb, bf16* __restrict__ O) {
  __shared__ bf16 Kl[2][32 * 72];     // [key][d], stride 72 (TDM pad 128B+16B)
  __shared__ bf16 Vt[2][64 * 40];     // [d][key], stride 40
  __shared__ bf16 Pl[8 * 16 * 40];    // per-wave P tile

  const int tid  = threadIdx.x;
  const int lane = tid & 31;
  const int w    = tid >> 5;
  const int lr   = lane & 15;
  const int sel  = lane >> 4;

  const int bh  = blockIdx.y;
  const int b   = bh >> 4;
  const int h   = bh & 15;
  const int hd0 = h * HEAD_DIM;
  const int q0  = blockIdx.x * 128 + w * 16;
  const size_t qrow = (size_t)(b * SEQ + q0);

  v16bf qa[2];
#pragma unroll
  for (int kb = 0; kb < 2; ++kb) {
    const bf16* p = Q + (qrow + lr) * D_MODEL + hd0 + kb * 32;
    qa[kb] = mk16(*(const v8bf*)(p + sel * 8), *(const v8bf*)(p + 16 + sel * 8));
  }

  v8f o[4] = {};
  float mrow[8], lrow[8];
#pragma unroll
  for (int i = 0; i < 8; ++i) { mrow[i] = -3.0e38f; lrow[i] = 0.f; }

  bf16* Pw = &Pl[w * 16 * 40];
  const float scl = 0.125f;  // 1/sqrt(64)

  const int kk = tid >> 3;          // key row 0..31
  const int d8 = (tid & 7) * 8;     // d offset 0..56

  const int NCH = SEQ / 32;

  // ---- prologue: stage chunk 0
  if (w == 0)
    tdm_load_2d((unsigned)(__UINTPTR_TYPE__)&Kl[0][0],
                &Kb[(size_t)(b * SEQ) * D_MODEL + hd0],
                64, 32, D_MODEL, 4u, 3u);
  {
    v8bf vv = *(const v8bf*)&Vb[(size_t)(b * SEQ + kk) * D_MODEL + hd0 + d8];
#pragma unroll
    for (int j = 0; j < 8; ++j) Vt[0][(d8 + j) * 40 + kk] = vv[j];
  }

  for (int ci = 0; ci < NCH; ++ci) {
    const int cur = ci & 1, nxt = cur ^ 1;
    if (w == 0) __builtin_amdgcn_s_wait_tensorcnt(0);
    __syncthreads();

    const bool pre = (ci + 1) < NCH;
    const int c0n  = (ci + 1) * 32;

    if (pre && w == 0)
      tdm_load_2d((unsigned)(__UINTPTR_TYPE__)&Kl[nxt][0],
                  &Kb[(size_t)(b * SEQ + c0n) * D_MODEL + hd0],
                  64, 32, D_MODEL, 4u, 3u);

    v8bf vv;
    if (pre)
      vv = *(const v8bf*)&Vb[(size_t)(b * SEQ + c0n + kk) * D_MODEL + hd0 + d8];

    // scores[16q x 32k] = Q @ K^T
    v8f sc[2] = {};
#pragma unroll
    for (int s = 0; s < 2; ++s)
#pragma unroll
      for (int kb = 0; kb < 2; ++kb) {
        const bf16* p = &Kl[cur][(s * 16 + lr) * 72 + kb * 32 + sel * 8];
        sc[s] = wmma_bf16(qa[kb], mk16(*(const v8bf*)p, *(const v8bf*)(p + 16)), sc[s]);
      }

    // online softmax (row stats live in each 16-lane half)
#pragma unroll
    for (int i = 0; i < 8; ++i) {
      float s0 = sc[0][i] * scl;
      float s1 = sc[1][i] * scl;
      float cm   = redmax16(fmaxf(s0, s1));
      float mnew = fmaxf(mrow[i], cm);
      float p0   = __expf(s0 - mnew);
      float p1   = __expf(s1 - mnew);
      float corr = __expf(mrow[i] - mnew);
      float ps   = redsum16(p0 + p1);
      lrow[i] = lrow[i] * corr + ps;
      mrow[i] = mnew;
#pragma unroll
      for (int j = 0; j < 4; ++j) o[j][i] *= corr;
      int prow = i + sel * 8;
      Pw[prow * 40 + lr]      = (bf16)p0;
      Pw[prow * 40 + 16 + lr] = (bf16)p1;
    }
    asm volatile("s_wait_dscnt 0x0" ::: "memory");  // wave-local P RAW fence

    // O += P @ V
    const bf16* pp = &Pw[lr * 40];
    v16bf pf = mk16(*(const v8bf*)(pp + sel * 8), *(const v8bf*)(pp + 16 + sel * 8));
#pragma unroll
    for (int j = 0; j < 4; ++j) {
      const bf16* vp = &Vt[cur][(j * 16 + lr) * 40 + sel * 8];
      o[j] = wmma_bf16(pf, mk16(*(const v8bf*)vp, *(const v8bf*)(vp + 16)), o[j]);
    }

    // commit next V tile (transposed)
    if (pre) {
#pragma unroll
      for (int j = 0; j < 8; ++j) Vt[nxt][(d8 + j) * 40 + kk] = vv[j];
    }
  }

  // normalize + write merged-head output
#pragma unroll
  for (int i = 0; i < 8; ++i) {
    float inv  = 1.0f / lrow[i];
    size_t row = qrow + i + sel * 8;
#pragma unroll
    for (int j = 0; j < 4; ++j)
      O[row * D_MODEL + hd0 + j * 16 + lr] = (bf16)(o[j][i] * inv);
  }
}

// ---------------------------------------------------------------------------
extern "C" void kernel_launch(void* const* d_in, const int* in_sizes, int n_in,
                              void* d_out, int out_size, void* d_ws, size_t ws_size,
                              hipStream_t stream) {
  (void)in_sizes; (void)n_in; (void)out_size; (void)ws_size;
  const float* x  = (const float*)d_in[0];
  const float* wq = (const float*)d_in[1];
  const float* wk = (const float*)d_in[2];
  const float* wv = (const float*)d_in[3];
  const float* wo = (const float*)d_in[4];

  char* p = (char*)d_ws;
  const size_t XB = (size_t)ROWS * D_MODEL * sizeof(bf16);
  const size_t WB = (size_t)D_MODEL * D_MODEL * sizeof(bf16);
  bf16* Xb  = (bf16*)p; p += XB;
  bf16* Wqb = (bf16*)p; p += WB;
  bf16* Wkb = (bf16*)p; p += WB;
  bf16* Wvb = (bf16*)p; p += WB;
  bf16* Wob = (bf16*)p; p += WB;
  bf16* Qb  = (bf16*)p; p += XB;
  bf16* Kbf = (bf16*)p; p += XB;
  bf16* Vbf = (bf16*)p; p += XB;
  bf16* Ob  = (bf16*)p; p += XB;   // ~48 MB total

  const int nx = ROWS * D_MODEL;
  const int nw = D_MODEL * D_MODEL;
  cvt_f32_bf16<<<nx / 1024, 256, 0, stream>>>(x,  Xb,  nx);
  cvt_f32_bf16<<<nw / 1024, 256, 0, stream>>>(wq, Wqb, nw);
  cvt_f32_bf16<<<nw / 1024, 256, 0, stream>>>(wk, Wkb, nw);
  cvt_f32_bf16<<<nw / 1024, 256, 0, stream>>>(wv, Wvb, nw);
  cvt_f32_bf16<<<nw / 1024, 256, 0, stream>>>(wo, Wob, nw);

  dim3 gg(D_MODEL / 128, ROWS / 128);  // (8, 32)
  gemm_bf16_kernel<false><<<gg, 256, 0, stream>>>(Xb, Wqb, Qb, nullptr, ROWS, D_MODEL, D_MODEL);
  gemm_bf16_kernel<false><<<gg, 256, 0, stream>>>(Xb, Wkb, Kbf, nullptr, ROWS, D_MODEL, D_MODEL);
  gemm_bf16_kernel<false><<<gg, 256, 0, stream>>>(Xb, Wvb, Vbf, nullptr, ROWS, D_MODEL, D_MODEL);

  attn_kernel<<<dim3(SEQ / 128, BATCH * NUM_HEADS), 256, 0, stream>>>(Qb, Kbf, Vbf, Ob);

  gemm_bf16_kernel<true><<<gg, 256, 0, stream>>>(Ob, Wob, nullptr, (float*)d_out, ROWS, D_MODEL, D_MODEL);
}
